// GloVe_33251636805826
// MI455X (gfx1250) — matrix operand order repaired and checked by reading, not compiled, not run on hardware
//
#include <hip/hip_runtime.h>
#include <hip/hip_bf16.h>
#include <math.h>

#define VOCAB 12288
#define EMB   128
#define TI    128              // tile rows (context / i)
#define TJ    64               // tile cols (target / j)
#define LDS_STRIDE 132         // bf16 elems per row: 128 + 4 pad (even -> dword-aligned pairs)

typedef __attribute__((ext_vector_type(16))) __bf16 v16bf;
typedef __attribute__((ext_vector_type(8)))  float  v8f;

union Frag {
    v16bf v;
    unsigned int u[8];
};

__device__ __forceinline__ unsigned short f32_to_bf16_rne(float x) {
    unsigned int b = __float_as_uint(x);
    unsigned int r = b + 0x7FFFu + ((b >> 16) & 1u);
    return (unsigned short)(r >> 16);
}
__device__ __forceinline__ float bf16_bits_to_f32(unsigned short h) {
    return __uint_as_float(((unsigned int)h) << 16);
}
// raw hardware transcendentals: v_log_f32 (base-2), v_exp_f32 (base-2)
__device__ __forceinline__ float fast_log2(float x) { return __builtin_amdgcn_logf(x); }
__device__ __forceinline__ float fast_exp2(float x) { return __builtin_amdgcn_exp2f(x); }

__global__ __launch_bounds__(256)
void glove_tile_kernel(const float* __restrict__ cooc,
                       const float* __restrict__ tgt,
                       const float* __restrict__ ctx,
                       const float* __restrict__ tbias,
                       const float* __restrict__ cbias,
                       float* __restrict__ partials,
                       float* __restrict__ out_atomic)
{
    __shared__ unsigned short sA_hi[TI * LDS_STRIDE];
    __shared__ unsigned short sA_lo[TI * LDS_STRIDE];
    __shared__ unsigned short sB_hi[TJ * LDS_STRIDE];
    __shared__ unsigned short sB_lo[TJ * LDS_STRIDE];
    __shared__ float s_cb[TI];
    __shared__ float s_tb[TJ];
    __shared__ float s_wavesum[8];

    const int tid = threadIdx.x;
    const int i0  = blockIdx.y * TI;
    const int j0  = blockIdx.x * TJ;

    // ---- stage biases ----
    if (tid < TI)            s_cb[tid]      = cbias[i0 + tid];
    else if (tid < TI + TJ)  s_tb[tid - TI] = tbias[j0 + (tid - TI)];

    // ---- stage context tile (A: 128 rows) as bf16 hi/lo splits ----
    for (int e = tid * 4; e < TI * EMB; e += 256 * 4) {
        const int r = e >> 7;       // e / 128
        const int c = e & 127;
        float4 xa = *(const float4*)(ctx + (size_t)(i0 + r) * EMB + c);
        float av[4] = {xa.x, xa.y, xa.z, xa.w};
        const int base = r * LDS_STRIDE + c;
        #pragma unroll
        for (int q = 0; q < 4; ++q) {
            unsigned short ah = f32_to_bf16_rne(av[q]);
            sA_hi[base + q] = ah;
            sA_lo[base + q] = f32_to_bf16_rne(av[q] - bf16_bits_to_f32(ah));
        }
    }
    // ---- stage target tile (B: 64 rows) ----
    for (int e = tid * 4; e < TJ * EMB; e += 256 * 4) {
        const int r = e >> 7;
        const int c = e & 127;
        float4 xb = *(const float4*)(tgt + (size_t)(j0 + r) * EMB + c);
        float bv[4] = {xb.x, xb.y, xb.z, xb.w};
        const int base = r * LDS_STRIDE + c;
        #pragma unroll
        for (int q = 0; q < 4; ++q) {
            unsigned short bh = f32_to_bf16_rne(bv[q]);
            sB_hi[base + q] = bh;
            sB_lo[base + q] = f32_to_bf16_rne(bv[q] - bf16_bits_to_f32(bh));
        }
    }
    __syncthreads();

    // ---- WMMA phase: each wave owns a 32(M) x 32(N) sub-tile = 2x2 WMMA tiles ----
    const int lane = tid & 31;
    const int wave = tid >> 5;
    const int l    = lane & 15;
    const int h    = lane >> 4;
    const int wm   = (wave & 3) * 32;   // M offset inside tile (4 wave rows)
    const int wn   = (wave >> 2) * 32;  // N offset inside tile (2 wave cols)

    v8f acc[2][2];
    #pragma unroll
    for (int a = 0; a < 2; ++a)
        #pragma unroll
        for (int b = 0; b < 2; ++b)
            acc[a][b] = v8f{};

    const unsigned int* __restrict__ pAh = (const unsigned int*)sA_hi;
    const unsigned int* __restrict__ pAl = (const unsigned int*)sA_lo;
    const unsigned int* __restrict__ pBh = (const unsigned int*)sB_hi;
    const unsigned int* __restrict__ pBl = (const unsigned int*)sB_lo;

    #pragma unroll
    for (int kb = 0; kb < EMB; kb += 32) {
        // A fragments (ISA layout: lane l = M row; VGPR v -> K = (v>=4?16:0) + 8h + 2(v&3))
        Frag aHi[2], aLo[2];
        #pragma unroll
        for (int sm = 0; sm < 2; ++sm) {
            const int row = wm + sm * 16 + l;
            #pragma unroll
            for (int v = 0; v < 8; ++v) {
                const int k   = kb + ((v & 4) << 2) + (h << 3) + ((v & 3) << 1);
                const int off = (row * LDS_STRIDE + k) >> 1;
                aHi[sm].u[v] = pAh[off];
                aLo[sm].u[v] = pAl[off];
            }
        }
        #pragma unroll
        for (int sn = 0; sn < 2; ++sn) {
            // B fragment (lane l = N col; VGPR v -> K = (v>=4?16:0) + 8h + 2(v&3))
            Frag bHi, bLo;
            const int col = wn + sn * 16 + l;
            #pragma unroll
            for (int v = 0; v < 8; ++v) {
                const int k   = kb + ((v & 4) << 2) + (h << 3) + ((v & 3) << 1);
                const int off = (col * LDS_STRIDE + k) >> 1;
                bHi.u[v] = pBh[off];
                bLo.u[v] = pBl[off];
            }
            #pragma unroll
            for (int sm = 0; sm < 2; ++sm) {
                // fp32-grade product via bf16x3: hi*hi + hi*lo + lo*hi
                acc[sm][sn] = __builtin_amdgcn_wmma_f32_16x16x32_bf16(
                    false, aHi[sm].v, false, bHi.v, (short)0, acc[sm][sn], false, false);
                acc[sm][sn] = __builtin_amdgcn_wmma_f32_16x16x32_bf16(
                    false, aHi[sm].v, false, bLo.v, (short)0, acc[sm][sn], false, false);
                acc[sm][sn] = __builtin_amdgcn_wmma_f32_16x16x32_bf16(
                    false, aLo[sm].v, false, bHi.v, (short)0, acc[sm][sn], false, false);
            }
        }
    }

    // ---- elementwise GloVe loss over this tile (stream X once, non-temporal) ----
    const float LN2 = 0.69314718055994531f;
    float local = 0.0f;
    #pragma unroll
    for (int sm = 0; sm < 2; ++sm) {
        #pragma unroll
        for (int sn = 0; sn < 2; ++sn) {
            #pragma unroll
            for (int r = 0; r < 8; ++r) {
                // C/D layout: VGPR r -> M = r (lanes 0-15) or r+8 (lanes 16-31), N = l
                const int mi = wm + sm * 16 + r + (h << 3);
                const int nj = wn + sn * 16 + l;
                const size_t gi = (size_t)(i0 + mi);
                const size_t gj = (size_t)(j0 + nj);
                const float X   = __builtin_nontemporal_load(cooc + gi * VOCAB + gj);
                const float dot = acc[sm][sn][r];
                // log1p(X) == ln2 * log2(1+X); X >= 0 so no cancellation issues
                const float d   = dot + s_tb[nj] + s_cb[mi] - LN2 * fast_log2(1.0f + X);
                // weight = min((X/100)^0.75, 1); log2(0) -> -inf -> exp2 -> 0 handles X==0
                const float w   = fminf(fast_exp2(0.75f * fast_log2(X * 0.01f)), 1.0f);
                local = fmaf(w * d, d, local);
            }
        }
    }

    // ---- reduce: wave shuffle -> LDS -> one value per block ----
    #pragma unroll
    for (int off = 16; off > 0; off >>= 1)
        local += __shfl_xor(local, off, 32);
    if (lane == 0) s_wavesum[wave] = local;
    __syncthreads();
    if (tid == 0) {
        float s = 0.0f;
        #pragma unroll
        for (int w = 0; w < 8; ++w) s += s_wavesum[w];
        if (partials) partials[blockIdx.y * gridDim.x + blockIdx.x] = s;
        else          atomicAdd(out_atomic, s);
    }
}

__global__ __launch_bounds__(256)
void glove_reduce_kernel(const float* __restrict__ partials, int n, float* __restrict__ out)
{
    __shared__ float sm[256];
    float s = 0.0f;
    for (int i = threadIdx.x; i < n; i += 256) s += partials[i];
    sm[threadIdx.x] = s;
    __syncthreads();
    for (int step = 128; step > 0; step >>= 1) {
        if (threadIdx.x < step) sm[threadIdx.x] += sm[threadIdx.x + step];
        __syncthreads();
    }
    if (threadIdx.x == 0) out[0] = sm[0];
}

extern "C" void kernel_launch(void* const* d_in, const int* in_sizes, int n_in,
                              void* d_out, int out_size, void* d_ws, size_t ws_size,
                              hipStream_t stream)
{
    const float* cooc = (const float*)d_in[0];   // matriz_coocurrencias [V,V]
    const float* tgt  = (const float*)d_in[1];   // target_emb  [V,EMB]
    const float* ctx  = (const float*)d_in[2];   // context_emb [V,EMB]
    const float* tb   = (const float*)d_in[3];   // target_bias  [V,1]
    const float* cb   = (const float*)d_in[4];   // context_bias [V,1]
    float* out = (float*)d_out;

    dim3 grid(VOCAB / TJ, VOCAB / TI);           // 192 x 96 tiles
    const int ntiles = (VOCAB / TI) * (VOCAB / TJ);

    if (ws_size >= (size_t)ntiles * sizeof(float)) {
        // deterministic two-stage reduction through workspace
        float* partials = (float*)d_ws;
        glove_tile_kernel<<<grid, 256, 0, stream>>>(cooc, tgt, ctx, tb, cb, partials, nullptr);
        glove_reduce_kernel<<<1, 256, 0, stream>>>(partials, ntiles, out);
    } else {
        hipMemsetAsync(out, 0, sizeof(float), stream);
        glove_tile_kernel<<<grid, 256, 0, stream>>>(cooc, tgt, ctx, tb, cb, nullptr, out);
    }
}
